// KOSLM_47528108098135
// MI455X (gfx1250) — matrix operand order repaired and checked by reading, not compiled, not run on hardware
//
#include <hip/hip_runtime.h>
#include <hip/hip_bf16.h>

// ---------------------------------------------------------------------------
// Model dims (fixed): B=8, L=512, D=1024, H=2048, NL=2
//   gates:  (4096,1024)@(1024,4096) -> precomputed for all t (WMMA)
//   scan t: h@Whh^T (8x2048x4096), innov@k1^T (8x2048x6144), r1@k2^T (8x6144x2048)
//   out:    (4096,2048)@(2048,1024) (WMMA)
// Scan GEMMs: one 16-col tile per block, K split across the block's 4 waves,
// partials reduced through LDS. K-loop is double-buffered with explicit
// sched_barriers so loads for chunk k+32 stay in flight across the WMMA on
// chunk k (verified: staged partial s_wait_loadcnt). NT=4 GEMMs fold the
// inter-tile stride into constant instruction offsets (FOLD) to avoid the
// base-pointer spill seen with 5 live pointer pairs. Weights converted to
// bf16 once per call (~67 MB weight stream per scan step, L2-resident).
// ---------------------------------------------------------------------------

typedef __attribute__((ext_vector_type(16))) __bf16 v16bf;
typedef __attribute__((ext_vector_type(8)))  float  v8f;
typedef __hip_bfloat16 bf16;

#if defined(__has_builtin)
#if __has_builtin(__builtin_amdgcn_sched_barrier)
#define SCHED_FENCE() __builtin_amdgcn_sched_barrier(0)
#endif
#endif
#ifndef SCHED_FENCE
#define SCHED_FENCE() ((void)0)
#endif

union BC32 { uint4 u[2]; v16bf v; };
static __device__ __forceinline__ v16bf pack16(uint4 lo, uint4 hi) {
  BC32 b; b.u[0] = lo; b.u[1] = hi; return b.v;
}

// Double-buffered K-loop over NT 16x16 output tiles (K must be a multiple of 64).
// A: bf16 activations, 16 rows x lda, row-major.
//    A-frag (16-bit 16x32): lanes 0-15 row=ln hold K {k0..+7, k0+16..+23},
//    lanes 16-31 same row hold K {k0+8..+15, k0+24..+31}.
// W: bf16 weights, row-major (N,K) == B^T -> per-lane B-frag is 2 contiguous b128:
//    lanes 0-15 col=ln hold K {k0..+7, k0+16..+23}, lanes 16-31 hold K {k0+8..+15, k0+24..+31}.
// FOLD: tiles at nrow[0] + j*16 rows -> single base pointer, j*32*LDW-byte
//       constant offsets (must stay within the signed 24-bit ioffset field).
template <int NT, bool PF, int LDW, bool FOLD>
static __device__ __forceinline__ void wmma_kloop(
    v8f (&acc)[NT],
    const bf16* __restrict__ A, int lda,
    const bf16* __restrict__ W,
    const int (&nrow)[NT], int K, int lane)
{
  const int sel = lane >> 4;
  const int ln  = lane & 15;
  const char* arow = (const char*)(A + (size_t)ln * lda);
  const char* brow[NT];
  brow[0] = (const char*)(W + (size_t)(nrow[0] + ln) * LDW);
  if (!FOLD) {
#pragma unroll
    for (int j = 1; j < NT; ++j)
      brow[j] = (const char*)(W + (size_t)(nrow[j] + ln) * LDW);
  }

  uint4 aL[2], aH[2], bL[2][NT], bH[2][NT];
  auto LD = [&](int s, int k0) {
    const int ab = 2 * (k0 + sel * 8);
    aL[s] = *(const uint4*)(arow + ab);
    aH[s] = *(const uint4*)(arow + ab + 32);
    const int bb = 2 * (k0 + sel * 8);
#pragma unroll
    for (int j = 0; j < NT; ++j) {
      const char* bp = FOLD ? (brow[0] + (size_t)j * (32u * (unsigned)LDW) + bb)
                            : (brow[j] + bb);
      bL[s][j] = *(const uint4*)bp;
      bH[s][j] = *(const uint4*)(bp + 32);
    }
  };
  auto MM = [&](int s) {
    v16bf av = pack16(aL[s], aH[s]);
#pragma unroll
    for (int j = 0; j < NT; ++j)
      acc[j] = __builtin_amdgcn_wmma_f32_16x16x32_bf16(
          false, av, false, pack16(bL[s][j], bH[s][j]), (short)0, acc[j],
          false, false);
  };

  LD(0, 0);
#pragma unroll 1
  for (int k0 = 0; k0 < K; k0 += 64) {
    LD(1, k0 + 32);                      // in flight across MM(0)
    if (PF) {
#pragma unroll
      for (int j = 0; j < NT; ++j)       // global_prefetch_b8 (GL2 warm)
        __builtin_prefetch((FOLD ? brow[0] + (size_t)j * (32u * (unsigned)LDW)
                                 : brow[j]) + 2 * k0 + 512, 0, 3);
    }
    SCHED_FENCE();                       // keep loads above the WMMA group
    MM(0);
    LD(0, k0 + 64);                      // in flight across MM(1); tail unused
    SCHED_FENCE();
    MM(1);
  }
}

// ---------------------------------------------------------------------------
__global__ void cvt_kernel(const float* __restrict__ in, bf16* __restrict__ out, long n) {
  long i = (long)blockIdx.x * blockDim.x + threadIdx.x;
  long st = (long)gridDim.x * blockDim.x;
  for (; i < n; i += st) out[i] = __float2bfloat16(in[i]);
}

__global__ void init_state_kernel(bf16* h, bf16* innov, bf16* r1, float* c,
                                  float* sumsq, int* cnt) {
  int i = blockIdx.x * blockDim.x + threadIdx.x;
  int st = gridDim.x * blockDim.x;
  bf16 z = __float2bfloat16(0.f);
  for (int k = i; k < 16 * 2048; k += st) { h[k] = z; innov[k] = z; }
  for (int k = i; k < 16 * 6144; k += st) r1[k] = z;
  for (int k = i; k < 8 * 2048; k += st) c[k] = 0.f;
  if (i < 8) sumsq[i] = 0.f;
  if (i == 0) *cnt = 0;
}

// rmsnorm over D=1024, one block per row of (B*L) rows, bf16 output
__global__ void rmsnorm_kernel(const float* __restrict__ x,
                               const float* __restrict__ w,
                               bf16* __restrict__ xn) {
  __shared__ float red[256];
  const int row = blockIdx.x;
  const float* xr = x + (size_t)row * 1024;
  float s = 0.f;
  for (int d = threadIdx.x; d < 1024; d += 256) { float v = xr[d]; s += v * v; }
  red[threadIdx.x] = s;
  __syncthreads();
  for (int o = 128; o > 0; o >>= 1) {
    if (threadIdx.x < o) red[threadIdx.x] += red[threadIdx.x + o];
    __syncthreads();
  }
  float scale = rsqrtf(red[0] * (1.f / 1024.f) + 1e-5f);
  for (int d = threadIdx.x; d < 1024; d += 256)
    xn[(size_t)row * 1024 + d] = __float2bfloat16(xr[d] * scale * w[d]);
}

// Gx[(t*8+b), n] = xn @ Wih^T + (bih+bhh) ; M=4096,K=1024,N=4096
__global__ void gemm_gx_kernel(const bf16* __restrict__ xn,
                               const bf16* __restrict__ Wih,
                               const float* __restrict__ bih,
                               const float* __restrict__ bhh,
                               float* __restrict__ Gx) {
  const int lane = threadIdx.x & 31, wv = threadIdx.x >> 5;
  const int m0 = blockIdx.x * 16;
  const int nbase = blockIdx.y * 256 + wv * 64;
  int nrow[4] = {nbase, nbase + 16, nbase + 32, nbase + 48};
  v8f acc[4];
#pragma unroll
  for (int j = 0; j < 4; ++j)
#pragma unroll
    for (int r = 0; r < 8; ++r) acc[j][r] = 0.f;
  wmma_kloop<4, false, 1024, true>(acc, xn + (size_t)m0 * 1024, 1024, Wih, nrow,
                                   1024, lane);
  const int sel = lane >> 4, ln = lane & 15;
#pragma unroll
  for (int j = 0; j < 4; ++j) {
    const int col = nrow[j] + ln;
    const float bias = bih[col] + bhh[col];
#pragma unroll
    for (int r = 0; r < 8; ++r) {
      int rg = m0 + sel * 8 + r;          // global row b*512+t
      int b = rg >> 9, tt = rg & 511;
      Gx[(size_t)((tt << 3) + b) * 4096 + col] = acc[j][r] + bias;
    }
  }
}

// Block-level K-split reduction: 4 waves' partial accumulators summed via LDS.
template <int NJ>
static __device__ __forceinline__ void block_reduce(v8f (&acc)[NJ], v8f* red,
                                                    int wv, int lane) {
#pragma unroll
  for (int j = 0; j < NJ; ++j) {
    red[wv * 32 + lane] = acc[j];
    __syncthreads();
    if (wv == 0)
      acc[j] = red[lane] + red[32 + lane] + red[64 + lane] + red[96 + lane];
    __syncthreads();
  }
}

// step A: gates = Gx[t] + h@Whh^T ; z/M/innov. 128 blocks x 4 waves (K-split).
__global__ void stepA_kernel(const float* __restrict__ Gx,
                             const bf16* __restrict__ hb,
                             const bf16* __restrict__ Whh,
                             const float* __restrict__ zb,
                             const float* __restrict__ mb,
                             const float* __restrict__ ll,
                             const float* __restrict__ c,
                             float* __restrict__ zs, float* __restrict__ Ms,
                             bf16* __restrict__ innov, int t) {
  __shared__ v8f red[128];
  const int lane = threadIdx.x & 31, wv = threadIdx.x >> 5;
  const int nz = blockIdx.x * 16;
  const int sel = lane >> 4, ln = lane & 15;
  int nrow[2] = {nz, nz + 2048};
  v8f acc[2];
#pragma unroll
  for (int j = 0; j < 2; ++j)
#pragma unroll
    for (int r = 0; r < 8; ++r)
      acc[j][r] = (wv == 0 && sel == 0)
                      ? Gx[(size_t)((t << 3) + r) * 4096 + nrow[j] + ln]
                      : 0.f;
  const int kofs = wv * 512;                       // K=2048 split 4 ways
  wmma_kloop<2, true, 2048, false>(acc, hb + kofs, 2048, Whh + kofs, nrow, 512,
                                   lane);
  block_reduce<2>(acc, red, wv, lane);
  if (wv == 0 && sel == 0) {
    const int col = nz + ln;
    const float zbv = zb[col], mbv = mb[col];
    const float Ab = -expf(ll[col]);
#pragma unroll
    for (int r = 0; r < 8; ++r) {
      float z = tanhf(acc[0][r] + zbv);
      float M = 1.5f * tanhf(acc[1][r] + mbv);
      float iv = z - M * (Ab * c[r * 2048 + col]);
      iv = fminf(fmaxf(iv, -20.f), 20.f);
      zs[r * 2048 + col] = z;
      Ms[r * 2048 + col] = M;
      innov[r * 2048 + col] = __float2bfloat16(iv);
    }
  }
}

// step B: r1 = gelu(innov@k1^T + k1_b). 384 blocks x 4 waves (K-split).
__global__ void stepB_kernel(const bf16* __restrict__ innov,
                             const bf16* __restrict__ K1,
                             const float* __restrict__ k1b,
                             bf16* __restrict__ r1) {
  __shared__ v8f red[128];
  const int lane = threadIdx.x & 31, wv = threadIdx.x >> 5;
  const int n0 = blockIdx.x * 16;
  const int sel = lane >> 4, ln = lane & 15;
  int nrow[1] = {n0};
  v8f acc[1];
#pragma unroll
  for (int r = 0; r < 8; ++r)
    acc[0][r] = (wv == 0 && sel == 0) ? k1b[n0 + ln] : 0.f;
  const int kofs = wv * 512;                       // K=2048 split 4 ways
  wmma_kloop<1, true, 2048, false>(acc, innov + kofs, 2048, K1 + kofs, nrow, 512,
                                   lane);
  block_reduce<1>(acc, red, wv, lane);
  if (wv == 0 && sel == 0) {
    const int col = n0 + ln;
#pragma unroll
    for (int r = 0; r < 8; ++r) {
      float v = acc[0][r];
      float g = 0.5f * v * (1.f + erff(v * 0.70710678f));  // exact gelu
      r1[r * 6144 + col] = __float2bfloat16(g);
    }
  }
}

// step C: rk = r1@k2^T + k2_b ; state update + c-norm (last-block finalize).
// 128 blocks x 4 waves (K-split of 6144).
__global__ void stepC_kernel(const bf16* __restrict__ r1,
                             const bf16* __restrict__ K2,
                             const float* __restrict__ k2b,
                             const float* __restrict__ ll,
                             const float* __restrict__ zs,
                             const float* __restrict__ Ms,
                             float* __restrict__ c, float* __restrict__ cpre,
                             bf16* __restrict__ hb, bf16* __restrict__ Hs,
                             float* __restrict__ sumsq, int* __restrict__ cnt,
                             int t) {
  __shared__ v8f red[128];
  __shared__ int lastFlag;
  const int lane = threadIdx.x & 31, wv = threadIdx.x >> 5;
  const int n0 = blockIdx.x * 16;
  const int sel = lane >> 4, ln = lane & 15;
  int nrow[1] = {n0};
  v8f acc[1];
#pragma unroll
  for (int r = 0; r < 8; ++r)
    acc[0][r] = (wv == 0 && sel == 0) ? k2b[n0 + ln] : 0.f;
  const int kofs = wv * 1536;                      // K=6144 split 4 ways
  wmma_kloop<1, true, 6144, false>(acc, r1 + kofs, 6144, K2 + kofs, nrow, 1536,
                                   lane);
  block_reduce<1>(acc, red, wv, lane);

  if (wv == 0) {
    float cn[8];
    const int col = n0 + ln;
    if (sel == 0) {
      const float Ab = -expf(ll[col]);
#pragma unroll
      for (int r = 0; r < 8; ++r) {
        float Kv = tanhf(acc[0][r]) * 0.45f;
        float M  = Ms[r * 2048 + col];
        M = fminf(fmaxf(M, -1.5f), 1.5f);
        float A = (1.f - Kv * M) * Ab;
        A = fminf(fmaxf(A, 0.f), 0.95f);
        float Bt = fminf(fmaxf(Kv, -0.95f), 0.95f);
        float cv = A * c[r * 2048 + col] + Bt * zs[r * 2048 + col];
        cpre[r * 2048 + col] = cv;
        cn[r] = cv;
      }
    } else {
#pragma unroll
      for (int r = 0; r < 8; ++r) cn[r] = 0.f;
    }
    // per-batch-row sum of squares: wave butterfly, then one atomic per row
#pragma unroll
    for (int r = 0; r < 8; ++r) {
      float v = cn[r] * cn[r];
      for (int m = 16; m > 0; m >>= 1) v += __shfl_xor(v, m, 32);
      if (lane == 0) atomicAdd(&sumsq[r], v);
    }
  }
  __syncthreads();
  if (threadIdx.x == 0) {
    __threadfence();
    int old = atomicAdd(cnt, 1);
    lastFlag = (old == (int)gridDim.x - 1);
  }
  __syncthreads();
  if (lastFlag) {
    __threadfence();
    for (int idx = threadIdx.x; idx < 8 * 2048; idx += blockDim.x) {
      int r = idx >> 11, n = idx & 2047;
      float ss = ((volatile float*)sumsq)[r];
      float nrm = sqrtf(ss * (1.f / 2048.f) + 1e-6f);
      float sc = fminf(100.f / nrm, 1.f);
      float cv = cpre[idx] * sc;
      c[idx] = cv;
      float M = Ms[idx];
      M = fminf(fmaxf(M, -1.5f), 1.5f);
      float h = M * cv;
      hb[idx] = __float2bfloat16(h);                        // next-step A operand
      Hs[(size_t)(r * 512 + t) * 2048 + n] = __float2bfloat16(h);
    }
    __syncthreads();
    if (threadIdx.x == 0) {
      for (int r = 0; r < 8; ++r) sumsq[r] = 0.f;
      *cnt = 0;
    }
  }
}

// y = Hs @ Wout^T ; M=4096,K=2048,N=1024
__global__ void gemm_out_kernel(const bf16* __restrict__ Hs,
                                const bf16* __restrict__ Wout,
                                float* __restrict__ y) {
  const int lane = threadIdx.x & 31, wv = threadIdx.x >> 5;
  const int m0 = blockIdx.x * 16;
  const int nbase = blockIdx.y * 256 + wv * 64;
  int nrow[4] = {nbase, nbase + 16, nbase + 32, nbase + 48};
  v8f acc[4];
#pragma unroll
  for (int j = 0; j < 4; ++j)
#pragma unroll
    for (int r = 0; r < 8; ++r) acc[j][r] = 0.f;
  wmma_kloop<4, false, 2048, true>(acc, Hs + (size_t)m0 * 2048, 2048, Wout, nrow,
                                   2048, lane);
  const int sel = lane >> 4, ln = lane & 15;
#pragma unroll
  for (int j = 0; j < 4; ++j) {
    const int col = nrow[j] + ln;
#pragma unroll
    for (int r = 0; r < 8; ++r) {
      int rg = m0 + sel * 8 + r;
      y[(size_t)rg * 1024 + col] = acc[j][r];
    }
  }
}

// ---------------------------------------------------------------------------
extern "C" void kernel_launch(void* const* d_in, const int* in_sizes, int n_in,
                              void* d_out, int out_size, void* d_ws, size_t ws_size,
                              hipStream_t stream) {
  (void)in_sizes; (void)n_in; (void)out_size; (void)ws_size;
  const float* x   = (const float*)d_in[0];
  const float* nw  = (const float*)d_in[1];
  const float* Wih = (const float*)d_in[2];
  const float* bih = (const float*)d_in[3];
  const float* Whh = (const float*)d_in[4];
  const float* bhh = (const float*)d_in[5];
  const float* ll  = (const float*)d_in[6];
  const float* mbp = (const float*)d_in[7];
  const float* zbp = (const float*)d_in[8];
  const float* k1w = (const float*)d_in[9];
  const float* k1b = (const float*)d_in[10];
  const float* k2w = (const float*)d_in[11];
  const float* k2b = (const float*)d_in[12];
  const float* Wo  = (const float*)d_in[13];

  char* ws = (char*)d_ws;
  size_t off = 0;
  auto alloc = [&](size_t bytes) -> void* {
    void* p = ws + off;
    off += (bytes + 255) & ~(size_t)255;
    return p;
  };
  bf16* wih_bf = (bf16*)alloc(2ull * 4096 * 1024 * 2);
  bf16* whh_bf = (bf16*)alloc(2ull * 4096 * 2048 * 2);
  bf16* k1_bf  = (bf16*)alloc(2ull * 6144 * 2048 * 2);
  bf16* k2_bf  = (bf16*)alloc(2ull * 2048 * 6144 * 2);
  bf16* wo_bf  = (bf16*)alloc(2ull * 1024 * 2048 * 2);
  bf16* xn_bf  = (bf16*)alloc(4096ull * 1024 * 2);
  float* Gx    = (float*)alloc(4096ull * 4096 * 4);
  bf16* Hs     = (bf16*)alloc(4096ull * 2048 * 2);
  float* ybuf  = (float*)alloc(4096ull * 1024 * 4);
  bf16* h_bf   = (bf16*)alloc(16ull * 2048 * 2);
  bf16* innov  = (bf16*)alloc(16ull * 2048 * 2);
  bf16* r1     = (bf16*)alloc(16ull * 6144 * 2);
  float* cbuf  = (float*)alloc(8ull * 2048 * 4);
  float* cpre  = (float*)alloc(8ull * 2048 * 4);
  float* zbuf  = (float*)alloc(8ull * 2048 * 4);
  float* Mbuf  = (float*)alloc(8ull * 2048 * 4);
  float* sumsq = (float*)alloc(8 * 4);
  int*   cnt   = (int*)alloc(4);

  // one-shot bf16 weight conversion (keeps scan weights L2-resident at half size)
  cvt_kernel<<<1024, 256, 0, stream>>>(Wih, wih_bf, 2ll * 4096 * 1024);
  cvt_kernel<<<1024, 256, 0, stream>>>(Whh, whh_bf, 2ll * 4096 * 2048);
  cvt_kernel<<<1024, 256, 0, stream>>>(k1w, k1_bf, 2ll * 6144 * 2048);
  cvt_kernel<<<1024, 256, 0, stream>>>(k2w, k2_bf, 2ll * 2048 * 6144);
  cvt_kernel<<<1024, 256, 0, stream>>>(Wo, wo_bf, 2ll * 1024 * 2048);

  for (int l = 0; l < 2; ++l) {
    const float* xin = (l == 0) ? x : ybuf;
    rmsnorm_kernel<<<4096, 256, 0, stream>>>(xin, nw, xn_bf);
    gemm_gx_kernel<<<dim3(256, 16), 128, 0, stream>>>(
        xn_bf, wih_bf + (size_t)l * 4096 * 1024, bih + l * 4096, bhh + l * 4096, Gx);
    init_state_kernel<<<96, 256, 0, stream>>>(h_bf, innov, r1, cbuf, sumsq, cnt);

    const bf16* whhl = whh_bf + (size_t)l * 4096 * 2048;
    const bf16* k1l  = k1_bf + (size_t)l * 6144 * 2048;
    const bf16* k2l  = k2_bf + (size_t)l * 2048 * 6144;
    const float* lll = ll + l * 2048;
    const float* mbl = mbp + l * 2048;
    const float* zbl = zbp + l * 2048;
    const float* k1bl = k1b + l * 6144;
    const float* k2bl = k2b + l * 2048;

    for (int t = 0; t < 512; ++t) {
      stepA_kernel<<<128, 128, 0, stream>>>(Gx, h_bf, whhl, zbl, mbl, lll, cbuf,
                                            zbuf, Mbuf, innov, t);
      stepB_kernel<<<384, 128, 0, stream>>>(innov, k1l, k1bl, r1);
      stepC_kernel<<<128, 128, 0, stream>>>(r1, k2l, k2bl, lll, zbuf, Mbuf, cbuf,
                                            cpre, h_bf, Hs, sumsq, cnt, t);
    }
    float* yout = (l == 0) ? ybuf : (float*)d_out;
    gemm_out_kernel<<<dim3(256, 4), 128, 0, stream>>>(
        Hs, wo_bf + (size_t)l * 1024 * 2048, yout);
  }
}
